// Feature_Fusion_Block_22179211116669
// MI455X (gfx1250) — matrix operand order repaired and checked by reading, not compiled, not run on hardware
//
#include <hip/hip_runtime.h>
#include <cstddef>

// ---------------------------------------------------------------------------
// Types for CDNA5 WMMA (wave32)
// ---------------------------------------------------------------------------
typedef __bf16 bf16_t;
typedef __attribute__((ext_vector_type(16))) __bf16 bf16x16;
typedef __attribute__((ext_vector_type(8)))  float  f32x8;

#define TM 64
#define TN 128
#define TK 32
#define LDS_STRIDE (TK + 8)   // 40 bf16 = 80 bytes, multiple of 16B -> aligned b128 frags

// ---------------------------------------------------------------------------
// WMMA bf16 GEMM: C[M,N] = A[M,K] * op(B) (+bias[n]) (+resid[m,n])
//   TRANSB = true : B is [N,K] row-major (weight layout, computes A @ B^T)
//   TRANSB = false: B is [K,N] row-major (computes A @ B)
//   EPI: 0 = plain, 1 = +bias, 2 = +bias +residual   (compile-time, no branches)
// fp32 inputs converted to bf16 on the global->LDS stage, f32 accumulate.
// Software-pipelined: double-buffered LDS, next tile's global loads are issued
// before the WMMAs so s_wait_loadcnt lands after the matrix ops.
// All dims must be multiples of the tile (true for every GEMM in this graph).
// ---------------------------------------------------------------------------
template<bool TRANSB, int EPI>
__global__ __launch_bounds__(256) void gemm_wmma_kernel(
    const float* __restrict__ A, int lda,
    const float* __restrict__ B, int ldb,
    const float* __restrict__ bias,
    const float* __restrict__ resid,
    float* __restrict__ C, int ldc, int K)
{
    __shared__ __align__(16) bf16_t As[2][TM][LDS_STRIDE];
    __shared__ __align__(16) bf16_t Bs[2][TN][LDS_STRIDE];

    const int tid  = threadIdx.x;
    const int lane = tid & 31;
    const int wave = tid >> 5;            // 0..7
    const int bm   = blockIdx.y * TM;
    const int bn   = blockIdx.x * TN;
    const int wm   = (wave >> 2) * 32;    // 0 or 32
    const int wn   = (wave & 3) * 32;     // 0,32,64,96

    f32x8 acc[2][2];
    #pragma unroll
    for (int i = 0; i < 2; ++i)
        #pragma unroll
        for (int j = 0; j < 2; ++j)
            acc[i][j] = f32x8{0.f,0.f,0.f,0.f,0.f,0.f,0.f,0.f};

    const int half = lane >> 4;   // 0 or 1
    const int r16  = lane & 15;

    // ---- per-thread staging coordinates (16B-aligned float4 loads)
    const int arow = tid >> 2;              // 0..63
    const int akk  = (tid & 3) * 8;         // 0,8,16,24
    const int brow = tid >> 1;              // TRANSB: 0..127
    const int bkk  = (tid & 1) * 16;        // TRANSB: 0,16
    const int bk   = tid >> 3;              // !TRANSB: 0..31
    const int bnn  = (tid & 7) * 16;        // !TRANSB: 0..112

    float4 aR0, aR1, bR0, bR1, bR2, bR3;

    auto load_tile = [&](int k0) {
        const float4* ap = reinterpret_cast<const float4*>(
            A + (size_t)(bm + arow) * lda + (k0 + akk));
        aR0 = ap[0]; aR1 = ap[1];
        if (TRANSB) {
            const float4* bp = reinterpret_cast<const float4*>(
                B + (size_t)(bn + brow) * ldb + (k0 + bkk));
            bR0 = bp[0]; bR1 = bp[1]; bR2 = bp[2]; bR3 = bp[3];
        } else {
            const float4* bp = reinterpret_cast<const float4*>(
                B + (size_t)(k0 + bk) * ldb + (bn + bnn));
            bR0 = bp[0]; bR1 = bp[1]; bR2 = bp[2]; bR3 = bp[3];
        }
    };
    auto store_tile = [&](int buf) {
        bf16_t* ad = &As[buf][arow][akk];
        ad[0] = (bf16_t)aR0.x; ad[1] = (bf16_t)aR0.y;
        ad[2] = (bf16_t)aR0.z; ad[3] = (bf16_t)aR0.w;
        ad[4] = (bf16_t)aR1.x; ad[5] = (bf16_t)aR1.y;
        ad[6] = (bf16_t)aR1.z; ad[7] = (bf16_t)aR1.w;
        float bs[16] = { bR0.x,bR0.y,bR0.z,bR0.w, bR1.x,bR1.y,bR1.z,bR1.w,
                         bR2.x,bR2.y,bR2.z,bR2.w, bR3.x,bR3.y,bR3.z,bR3.w };
        if (TRANSB) {
            bf16_t* bd = &Bs[buf][brow][bkk];
            #pragma unroll
            for (int j = 0; j < 16; ++j) bd[j] = (bf16_t)bs[j];
        } else {
            #pragma unroll
            for (int j = 0; j < 16; ++j) Bs[buf][bnn + j][bk] = (bf16_t)bs[j];
        }
    };

    const int NT = K / TK;
    load_tile(0);
    store_tile(0);
    __syncthreads();

    for (int kt = 0; kt < NT; ++kt) {
        const int cur = kt & 1;
        if (kt + 1 < NT) load_tile((kt + 1) * TK);   // issue early, wait later

        // ---- fragments per documented VGPR layouts (two b128 each)
        bf16x16 aF[2], bF[2];
        #pragma unroll
        for (int mi = 0; mi < 2; ++mi) {
            const bf16_t* p = &As[cur][wm + mi * 16 + r16][half * 8];
            union { bf16x16 v; uint4 q[2]; } u;
            u.q[0] = *(const uint4*)(p);
            u.q[1] = *(const uint4*)(p + 16);
            aF[mi] = u.v;
        }
        #pragma unroll
        for (int ni = 0; ni < 2; ++ni) {
            const bf16_t* p = &Bs[cur][wn + ni * 16 + r16][half * 8];
            union { bf16x16 v; uint4 q[2]; } u;
            u.q[0] = *(const uint4*)(p);
            u.q[1] = *(const uint4*)(p + 16);
            bF[ni] = u.v;
        }

        #pragma unroll
        for (int mi = 0; mi < 2; ++mi)
            #pragma unroll
            for (int ni = 0; ni < 2; ++ni)
                acc[mi][ni] = __builtin_amdgcn_wmma_f32_16x16x32_bf16(
                    false, aF[mi], false, bF[ni], (short)0, acc[mi][ni],
                    false, false);

        if (kt + 1 < NT) store_tile(cur ^ 1);        // waits loads after WMMAs
        __syncthreads();
    }

    // ---- epilogue: D element r -> row = base + 8*half + r, col = lane&15
    #pragma unroll
    for (int mi = 0; mi < 2; ++mi)
        #pragma unroll
        for (int ni = 0; ni < 2; ++ni) {
            const int col = bn + wn + ni * 16 + r16;
            float badd = 0.f;
            if (EPI >= 1) badd = bias[col];
            #pragma unroll
            for (int r = 0; r < 8; ++r) {
                const int row = bm + wm + mi * 16 + half * 8 + r;
                float v = acc[mi][ni][r] + badd;
                if (EPI == 2) v += resid[(size_t)row * ldc + col];
                C[(size_t)row * ldc + col] = v;
            }
        }
}

// ---------------------------------------------------------------------------
// softmax over last dim (contiguous rows), one wave per row
// ---------------------------------------------------------------------------
__global__ __launch_bounds__(256) void softmax_rows_kernel(
    const float* __restrict__ X, float* __restrict__ Y, int rows, int C)
{
    const int row = blockIdx.x * 8 + (threadIdx.x >> 5);
    if (row >= rows) return;
    const int lane = threadIdx.x & 31;
    const float* x = X + (size_t)row * C;
    float m = -3.4e38f;
    for (int c = lane; c < C; c += 32) m = fmaxf(m, x[c]);
    #pragma unroll
    for (int o = 16; o; o >>= 1) m = fmaxf(m, __shfl_xor(m, o, 32));
    float s = 0.f;
    for (int c = lane; c < C; c += 32) s += __expf(x[c] - m);
    #pragma unroll
    for (int o = 16; o; o >>= 1) s += __shfl_xor(s, o, 32);
    const float inv = 1.f / s;
    float* y = Y + (size_t)row * C;
    for (int c = lane; c < C; c += 32) y[c] = __expf(x[c] - m) * inv;
}

// ---------------------------------------------------------------------------
// softmax over dim-1 of [M,C]: block = 32 columns x 8 M-segments.
// Each wave covers one segment x 32 contiguous columns (coalesced); the
// 8 segment-partials are combined through LDS.
// ---------------------------------------------------------------------------
__global__ __launch_bounds__(256) void softmax_cols_kernel(
    const float* __restrict__ X, float* __restrict__ Y, int M, int C)
{
    __shared__ float red[8][32];
    const int lc  = threadIdx.x & 31;
    const int seg = threadIdx.x >> 5;            // 0..7
    const int col = blockIdx.x * 32 + lc;

    float m = -3.4e38f;
    for (int i = seg; i < M; i += 8) m = fmaxf(m, X[(size_t)i * C + col]);
    red[seg][lc] = m;
    __syncthreads();
    #pragma unroll
    for (int s2 = 0; s2 < 8; ++s2) m = fmaxf(m, red[s2][lc]);
    __syncthreads();

    float s = 0.f;
    for (int i = seg; i < M; i += 8) s += __expf(X[(size_t)i * C + col] - m);
    red[seg][lc] = s;
    __syncthreads();
    s = 0.f;
    #pragma unroll
    for (int s2 = 0; s2 < 8; ++s2) s += red[s2][lc];
    const float inv = 1.f / s;

    for (int i = seg; i < M; i += 8)
        Y[(size_t)i * C + col] = __expf(X[(size_t)i * C + col] - m) * inv;
}

// ---------------------------------------------------------------------------
// LayerNorm over rows of [rows,C]; optional elementwise add of X2 first,
// optional exact-erf GELU afterwards. One wave per row.
// ---------------------------------------------------------------------------
__global__ __launch_bounds__(256) void layernorm_kernel(
    const float* __restrict__ X, const float* __restrict__ X2,
    const float* __restrict__ g, const float* __restrict__ bta,
    float* __restrict__ Y, int rows, int C, int do_gelu)
{
    const int row = blockIdx.x * 8 + (threadIdx.x >> 5);
    if (row >= rows) return;
    const int lane = threadIdx.x & 31;
    const float* xr  = X + (size_t)row * C;
    const float* x2r = X2 ? X2 + (size_t)row * C : nullptr;
    float s = 0.f, s2 = 0.f;
    for (int c = lane; c < C; c += 32) {
        float v = xr[c] + (x2r ? x2r[c] : 0.f);
        s += v; s2 += v * v;
    }
    #pragma unroll
    for (int o = 16; o; o >>= 1) { s += __shfl_xor(s, o, 32); s2 += __shfl_xor(s2, o, 32); }
    const float mean = s / (float)C;
    const float var  = s2 / (float)C - mean * mean;
    const float inv  = rsqrtf(var + 1e-5f);
    float* yr = Y + (size_t)row * C;
    for (int c = lane; c < C; c += 32) {
        float v = xr[c] + (x2r ? x2r[c] : 0.f);
        float t = (v - mean) * inv * g[c] + bta[c];
        if (do_gelu) t = 0.5f * t * (1.f + erff(t * 0.70710678118654752f));
        yr[c] = t;
    }
}

// ---------------------------------------------------------------------------
// atten = x3 + f*a3 + o*a2 + o*a1, f = sigmoid(w), o = (1-f)/2
// ---------------------------------------------------------------------------
__global__ void fuse_kernel(const float* __restrict__ x3,
                            const float* __restrict__ a1,
                            const float* __restrict__ a2,
                            const float* __restrict__ a3,
                            const float* __restrict__ w,
                            float* __restrict__ atten, int n)
{
    const int i = blockIdx.x * blockDim.x + threadIdx.x;
    if (i >= n) return;
    const float f = 1.f / (1.f + __expf(-w[0]));
    const float o = (1.f - f) * 0.5f;
    atten[i] = x3[i] + f * a3[i] + o * a2[i] + o * a1[i];
}

// ---------------------------------------------------------------------------
// depthwise 3x3 SAME conv on h:[N=H*W, C] (token-major), OIHW weights [C,1,3,3]
// ---------------------------------------------------------------------------
__global__ void dwconv3x3_kernel(const float* __restrict__ H,
                                 const float* __restrict__ Wdw,
                                 const float* __restrict__ bdw,
                                 float* __restrict__ Out,
                                 int HH, int WW, int C)
{
    const int idx = blockIdx.x * blockDim.x + threadIdx.x;
    const int total = HH * WW * C;
    if (idx >= total) return;
    const int c = idx % C;
    const int n = idx / C;
    const int x = n % WW;
    const int y = n / WW;
    float acc = bdw[c];
    #pragma unroll
    for (int dy = -1; dy <= 1; ++dy) {
        const int yy = y + dy;
        if (yy < 0 || yy >= HH) continue;
        #pragma unroll
        for (int dx = -1; dx <= 1; ++dx) {
            const int xx = x + dx;
            if (xx < 0 || xx >= WW) continue;
            acc += H[((size_t)(yy * WW + xx)) * C + c] *
                   Wdw[c * 9 + (dy + 1) * 3 + (dx + 1)];
        }
    }
    Out[idx] = acc;
}

// ---------------------------------------------------------------------------
// Host-side helpers
// ---------------------------------------------------------------------------
static inline void launch_gemm(hipStream_t s, bool transB, int epi,
    const float* A, int lda, const float* B, int ldb,
    const float* bias, const float* resid,
    float* C, int ldc, int M, int N, int K)
{
    dim3 grid(N / TN, M / TM), block(256);
    if (transB) {
        if (epi == 0)
            gemm_wmma_kernel<true, 0><<<grid, block, 0, s>>>(A, lda, B, ldb, bias, resid, C, ldc, K);
        else if (epi == 1)
            gemm_wmma_kernel<true, 1><<<grid, block, 0, s>>>(A, lda, B, ldb, bias, resid, C, ldc, K);
        else
            gemm_wmma_kernel<true, 2><<<grid, block, 0, s>>>(A, lda, B, ldb, bias, resid, C, ldc, K);
    } else {
        gemm_wmma_kernel<false, 0><<<grid, block, 0, s>>>(A, lda, B, ldb, bias, resid, C, ldc, K);
    }
}

static inline void launch_softmax_rows(hipStream_t s, const float* X, float* Y, int rows, int C)
{
    softmax_rows_kernel<<<dim3((rows + 7) / 8), dim3(256), 0, s>>>(X, Y, rows, C);
}

static inline void launch_softmax_cols(hipStream_t s, const float* X, float* Y, int M, int C)
{
    softmax_cols_kernel<<<dim3(C / 32), dim3(256), 0, s>>>(X, Y, M, C);
}

static inline void launch_ln(hipStream_t s, const float* X, const float* X2,
                             const float* g, const float* b, float* Y,
                             int rows, int C, int gelu)
{
    layernorm_kernel<<<dim3((rows + 7) / 8), dim3(256), 0, s>>>(X, X2, g, b, Y, rows, C, gelu);
}

// ---------------------------------------------------------------------------
// kernel_launch
// ---------------------------------------------------------------------------
extern "C" void kernel_launch(void* const* d_in, const int* /*in_sizes*/, int /*n_in*/,
                              void* d_out, int /*out_size*/, void* d_ws, size_t /*ws_size*/,
                              hipStream_t stream)
{
    const float* X      = (const float*)d_in[0];   // [B,N1,d0]
    const float* X2     = (const float*)d_in[1];   // [B,N2,d1]
    const float* X3     = (const float*)d_in[2];   // [B,N3,d2]
    const float* W_lin  = (const float*)d_in[3];   // [d2,d0]
    const float* W_lin2 = (const float*)d_in[4];   // [d2,d1]
    const float* W_lin3 = (const float*)d_in[5];   // [d1,d0]
    const float* W_lin4 = (const float*)d_in[6];   // [d2,d1]
    const float* wgate  = (const float*)d_in[7];   // [1]
    const float* Wp1    = (const float*)d_in[8];
    const float* bp1    = (const float*)d_in[9];
    const float* Wp2    = (const float*)d_in[10];
    const float* bp2    = (const float*)d_in[11];
    const float* Wp3    = (const float*)d_in[12];
    const float* bp3    = (const float*)d_in[13];
    const float* Wp4    = (const float*)d_in[14];
    const float* bp4    = (const float*)d_in[15];
    const float* g_norm = (const float*)d_in[16];
    const float* b_norm = (const float*)d_in[17];
    const float* Wfc1   = (const float*)d_in[18];  // [dh,d2]
    const float* bfc1   = (const float*)d_in[19];
    const float* Wdw    = (const float*)d_in[20];  // [dh,1,3,3]
    const float* bdw    = (const float*)d_in[21];
    const float* g_ln1  = (const float*)d_in[22];
    const float* b_ln1  = (const float*)d_in[23];
    const float* Wfc2   = (const float*)d_in[24];  // [d2,dh]
    const float* bfc2   = (const float*)d_in[25];
    float* OUT = (float*)d_out;

    const int B = 4, N1 = 9216, N2 = 2304, N3 = 576;
    const int d0 = 128, d1 = 256, d2 = 512, dh = 2048;
    const int HSZ = 24, WSZ = 24;

    // per-batch workspace carve (reused every batch; stream serializes)
    float* ws = (float*)d_ws;
    size_t off = 0;
    auto carve = [&](size_t n) { float* p = ws + off; off += n; return p; };
    float* bufG   = carve((size_t)N1 * d2);   // gemm out pre-softmax (largest use)
    float* bufK   = carve((size_t)N1 * d2);   // softmax over C
    float* bufQ   = carve((size_t)N1 * d2);   // softmax over tokens
    float* bufCtx = carve((size_t)N2 * N1);   // largest ctx [2304,9216]
    float* bufT0  = carve((size_t)N2 * d2);   // att intermediates
    float* bufT1  = carve((size_t)N2 * d2);
    float* bufA3  = carve((size_t)N2 * d2);   // a3 [2304,512]
    float* bufK2  = carve((size_t)N2 * d2);
    float* bufQ2  = carve((size_t)N2 * d2);
    float* bufAt1 = carve((size_t)N3 * d2);
    float* bufAt2 = carve((size_t)N3 * d2);
    float* bufAt3 = carve((size_t)N3 * d2);
    float* bufAtt = carve((size_t)N3 * d2);   // atten (residual for final out)
    float* bufAn  = carve((size_t)N3 * d2);
    float* bufH   = carve((size_t)N3 * dh);
    float* bufDw  = carve((size_t)N3 * dh);
    float* bufAx  = carve((size_t)N3 * dh);

    for (int b = 0; b < B; ++b) {
        const float* x  = X  + (size_t)b * N1 * d0;
        const float* x2 = X2 + (size_t)b * N2 * d1;
        const float* x3 = X3 + (size_t)b * N3 * d2;
        float* out_b = OUT + (size_t)b * N3 * d2;

        // ---------------- path 3: a3 = proj(crossattn(x2, x@Wlin3^T)) @ Wlin4^T
        launch_gemm(stream, true, 0, x, d0, W_lin3, d0, nullptr, nullptr, bufG, d1, N1, d1, d0);
        launch_softmax_rows(stream, bufG, bufK, N1, d1);
        launch_softmax_cols(stream, bufG, bufQ, N1, d1);
        launch_gemm(stream, true,  0, x2, d1, bufK, d1, nullptr, nullptr, bufCtx, N1, N2, N1, d1); // ctx [2304,9216]
        launch_gemm(stream, false, 0, bufCtx, N1, bufQ, d1, nullptr, nullptr, bufT0, d1, N2, d1, N1);
        launch_gemm(stream, true,  1, bufT0, d1, Wp3, d1, bp3, nullptr, bufT1, d1, N2, d1, d1);
        launch_gemm(stream, true,  0, bufT1, d1, W_lin4, d1, nullptr, nullptr, bufA3, d2, N2, d2, d1);
        // atten3 = crossattn(x3, a3) @ Wp4^T + bp4
        launch_softmax_rows(stream, bufA3, bufK2, N2, d2);
        launch_softmax_cols(stream, bufA3, bufQ2, N2, d2);
        launch_gemm(stream, true,  0, x3, d2, bufK2, d2, nullptr, nullptr, bufCtx, N2, N3, N2, d2); // ctx [576,2304]
        launch_gemm(stream, false, 0, bufCtx, N2, bufQ2, d2, nullptr, nullptr, bufT0, d2, N3, d2, N2);
        launch_gemm(stream, true,  1, bufT0, d2, Wp4, d2, bp4, nullptr, bufAt3, d2, N3, d2, d2);

        // ---------------- path 1: atten1 = crossattn(x3, x@Wlin^T) @ Wp1^T + bp1
        launch_gemm(stream, true, 0, x, d0, W_lin, d0, nullptr, nullptr, bufG, d2, N1, d2, d0);
        launch_softmax_rows(stream, bufG, bufK, N1, d2);
        launch_softmax_cols(stream, bufG, bufQ, N1, d2);
        launch_gemm(stream, true,  0, x3, d2, bufK, d2, nullptr, nullptr, bufCtx, N1, N3, N1, d2); // ctx [576,9216]
        launch_gemm(stream, false, 0, bufCtx, N1, bufQ, d2, nullptr, nullptr, bufT0, d2, N3, d2, N1);
        launch_gemm(stream, true,  1, bufT0, d2, Wp1, d2, bp1, nullptr, bufAt1, d2, N3, d2, d2);

        // ---------------- path 2: atten2 = crossattn(x3, x2@Wlin2^T) @ Wp2^T + bp2
        launch_gemm(stream, true, 0, x2, d1, W_lin2, d1, nullptr, nullptr, bufG, d2, N2, d2, d1);
        launch_softmax_rows(stream, bufG, bufK, N2, d2);
        launch_softmax_cols(stream, bufG, bufQ, N2, d2);
        launch_gemm(stream, true,  0, x3, d2, bufK, d2, nullptr, nullptr, bufCtx, N2, N3, N2, d2); // ctx [576,2304]
        launch_gemm(stream, false, 0, bufCtx, N2, bufQ, d2, nullptr, nullptr, bufT0, d2, N3, d2, N2);
        launch_gemm(stream, true,  1, bufT0, d2, Wp2, d2, bp2, nullptr, bufAt2, d2, N3, d2, d2);

        // ---------------- fusion + LN + MixFFN
        {
            const int n = N3 * d2;
            fuse_kernel<<<dim3((n + 255) / 256), dim3(256), 0, stream>>>(
                x3, bufAt1, bufAt2, bufAt3, wgate, bufAtt, n);
        }
        launch_ln(stream, bufAtt, nullptr, g_norm, b_norm, bufAn, N3, d2, 0);
        launch_gemm(stream, true, 1, bufAn, d2, Wfc1, d2, bfc1, nullptr, bufH, dh, N3, dh, d2);
        {
            const int total = N3 * dh;
            dwconv3x3_kernel<<<dim3((total + 255) / 256), dim3(256), 0, stream>>>(
                bufH, Wdw, bdw, bufDw, HSZ, WSZ, dh);
        }
        launch_ln(stream, bufDw, bufH, g_ln1, b_ln1, bufAx, N3, dh, 1);
        // out = atten + ax @ Wfc2^T + bfc2   (residual fused in epilogue)
        launch_gemm(stream, true, 2, bufAx, dh, Wfc2, dh, bfc2, bufAtt, out_b, d2, N3, d2, dh);
    }
}
